// MGHCT_60017872995082
// MI455X (gfx1250) — compile-verified, matrix operand order checked by reading, not run on hardware
//
#include <hip/hip_runtime.h>
#include <hip/hip_bf16.h>
#include <math.h>

// Problem constants (from reference)
static constexpr int kN   = 10000;
static constexpr int kE   = 320000;
static constexpr int kCin = 512;
static constexpr int kC   = 128;
static constexpr int kK   = 8;
static constexpr int kD   = kC * kK;   // 1024
static constexpr int kL   = 8;
static constexpr float kSlope = 0.01f;

typedef __attribute__((ext_vector_type(16))) __bf16 v16bf;
typedef __attribute__((ext_vector_type(8)))  float  v8f;

union Frag16 { v16bf v; unsigned int u[8]; };

__device__ __forceinline__ unsigned short f2bf(float f) {
    unsigned int u = __float_as_uint(f);
    unsigned int r = u + 0x7FFFu + ((u >> 16) & 1u);   // round-to-nearest-even
    return (unsigned short)(r >> 16);
}

__device__ __forceinline__ float leaky2(float x) {
    // leaky(leaky(x)) with slope 0.01
    return (x >= 0.0f) ? x : (kSlope * kSlope) * x;
}

// ---------------------------------------------------------------------------
// Elementwise / setup kernels
// ---------------------------------------------------------------------------
__global__ void zero_f32(float* __restrict__ p, long long n) {
    long long i = (long long)blockIdx.x * blockDim.x + threadIdx.x;
    if (i < n) p[i] = 0.0f;
}

__global__ void deg_kernel(const int* __restrict__ dst, float* __restrict__ deg) {
    int e = blockIdx.x * blockDim.x + threadIdx.x;
    if (e < kE) atomicAdd(&deg[dst[e]], 1.0f);
}

__global__ void norm_kernel(const float* __restrict__ deg,
                            float* __restrict__ dinv, float* __restrict__ snorm) {
    int i = blockIdx.x * blockDim.x + threadIdx.x;
    if (i < kN) {
        float d = deg[i] + 1.0f;      // self-loop
        dinv[i]  = rsqrtf(d);
        snorm[i] = 1.0f / d;
    }
}

__global__ void enorm_kernel(const int* __restrict__ src, const int* __restrict__ dst,
                             const float* __restrict__ dinv, float* __restrict__ enorm) {
    int e = blockIdx.x * blockDim.x + threadIdx.x;
    if (e < kE) enorm[e] = dinv[src[e]] * dinv[dst[e]];
}

__global__ void cvt_bf16(const float* __restrict__ in, unsigned short* __restrict__ out,
                         long long n) {
    long long i = (long long)blockIdx.x * blockDim.x + threadIdx.x;
    if (i < n) out[i] = f2bf(in[i]);
}

// WcatT[j, f] = W_init[k, f, d]  (j = k*C + d), shape [D x Cin] row-major
__global__ void build_wcatT(const float* __restrict__ W, unsigned short* __restrict__ T) {
    long long i = (long long)blockIdx.x * blockDim.x + threadIdx.x;
    if (i >= (long long)kD * kCin) return;
    int f = (int)(i % kCin);
    int j = (int)(i / kCin);
    int k = j / kC, d = j % kC;
    T[i] = f2bf(W[((long long)k * kCin + f) * kC + d]);
}

// WconvT[(l*K+k)][d, c] = W_conv[l, k, c, d], per-batch [C x C] row-major
__global__ void build_wconvT(const float* __restrict__ W, unsigned short* __restrict__ T) {
    long long i = (long long)blockIdx.x * blockDim.x + threadIdx.x;
    if (i >= (long long)(kL - 1) * kK * kC * kC) return;
    int c  = (int)(i % kC);
    long long r = i / kC;
    int d  = (int)(r % kC);
    long long kb = r / kC;                     // l*K + k
    T[i] = f2bf(W[(kb * kC + c) * kC + d]);
}

// WresT[l][j, k2] = W_res[l][k2, j], per-layer [D x D] row-major
__global__ void build_wresT(const float* __restrict__ W, unsigned short* __restrict__ T) {
    long long i = (long long)blockIdx.x * blockDim.x + threadIdx.x;
    if (i >= (long long)(kL - 1) * kD * kD) return;
    int k2 = (int)(i % kD);
    long long r = i / kD;
    int j  = (int)(r % kD);
    long long l = r / kD;
    T[i] = f2bf(W[(l * kD + k2) * kD + j]);
}

// ---------------------------------------------------------------------------
// WMMA bf16 GEMM:  C[M x Ncols] = A[M x Kdim] * B[Kdim x Ncols]
// B supplied pre-transposed (Bt is [Ncols x Kdim] row-major) so A and B
// fragments share the 16-bit A-matrix VGPR layout (CDNA5 ISA 7.12.2).
// One wave -> one 16x64 output strip (4 N-tiles): the A fragment is loaded
// once per K-step and reused by 4 consecutive WMMAs (1 A + 4 B fragments
// per 4 WMMAs instead of 4+4), cutting fragment load traffic by 37.5%.
// wavesPerBlock = blockDim.x/32 strips per block along N.
// ---------------------------------------------------------------------------
__device__ __forceinline__ v16bf load_frag16x32(const unsigned short* __restrict__ tile,
                                                int ld, int kbase, int lane) {
    Frag16 f;
    const int half = (lane >> 4) & 1;
    const int row  = lane & 15;
    const unsigned short* rp = tile + (long long)row * ld + kbase + half * 8;
#pragma unroll
    for (int v = 0; v < 8; ++v) {
        int k = ((v < 4) ? 0 : 16) + (v & 3) * 2;   // pairs (k, k+1) -> one dword
        f.u[v] = *(const unsigned int*)(rp + k);
    }
    return f.v;
}

__global__ void wmma_gemm_bf16(const unsigned short* __restrict__ A, int lda, long long strideA,
                               const unsigned short* __restrict__ Bt, int ldb, long long strideB,
                               float* __restrict__ C, int ldc, long long strideC,
                               int Kdim) {
    const int lane  = threadIdx.x & 31;
    const int wave  = threadIdx.x >> 5;
    const int wpb   = blockDim.x >> 5;
    const int mt    = blockIdx.x;
    const int strip = blockIdx.y * wpb + wave;      // 64-column strip index
    const long long b = blockIdx.z;

    const unsigned short* Abase = A  + b * strideA + (long long)(mt * 16) * lda;
    const unsigned short* Bbase = Bt + b * strideB + (long long)(strip * 64) * ldb;

    v8f acc0 = {}, acc1 = {}, acc2 = {}, acc3 = {};
    for (int k = 0; k < Kdim; k += 32) {
        v16bf af = load_frag16x32(Abase, lda, k, lane);
        v16bf b0 = load_frag16x32(Bbase,                          ldb, k, lane);
        v16bf b1 = load_frag16x32(Bbase + (long long)16 * ldb,    ldb, k, lane);
        v16bf b2 = load_frag16x32(Bbase + (long long)32 * ldb,    ldb, k, lane);
        v16bf b3 = load_frag16x32(Bbase + (long long)48 * ldb,    ldb, k, lane);
        acc0 = __builtin_amdgcn_wmma_f32_16x16x32_bf16(false, af, false, b0, (short)0, acc0, false, false);
        acc1 = __builtin_amdgcn_wmma_f32_16x16x32_bf16(false, af, false, b1, (short)0, acc1, false, false);
        acc2 = __builtin_amdgcn_wmma_f32_16x16x32_bf16(false, af, false, b2, (short)0, acc2, false, false);
        acc3 = __builtin_amdgcn_wmma_f32_16x16x32_bf16(false, af, false, b3, (short)0, acc3, false, false);
    }

    const int half = (lane >> 4) & 1;
    float* Cp = C + b * strideC
                  + (long long)(mt * 16 + half * 8) * ldc
                  + strip * 64 + (lane & 15);
#pragma unroll
    for (int r = 0; r < 8; ++r) {
        long long ro = (long long)r * ldc;
        Cp[ro]      = acc0[r];
        Cp[ro + 16] = acc1[r];
        Cp[ro + 32] = acc2[r];
        Cp[ro + 48] = acc3[r];
    }
}

// ---------------------------------------------------------------------------
// Graph aggregation: acc[dst] += feat[src] * enorm[e].  One block per edge,
// 256 lanes x float4 covers D=1024.  acc/feat are L2-resident (41 MB << 192 MB).
// ---------------------------------------------------------------------------
__global__ void scatter_kernel(const int* __restrict__ src, const int* __restrict__ dst,
                               const float* __restrict__ enorm,
                               const float* __restrict__ feat, float* __restrict__ acc) {
    int e = blockIdx.x;
    float w = enorm[e];
    int s = src[e], d = dst[e];
    const float4* fp = (const float4*)(feat + (long long)s * kD);
    float* ap = acc + (long long)d * kD;
    int j = threadIdx.x;                       // 256 threads, D/4 = 256 float4s
    float4 v = fp[j];
    atomicAdd(ap + 4 * j + 0, v.x * w);
    atomicAdd(ap + 4 * j + 1, v.y * w);
    atomicAdd(ap + 4 * j + 2, v.z * w);
    atomicAdd(ap + 4 * j + 3, v.w * w);
}

// ---------------------------------------------------------------------------
// Combine kernels. gate == identity (softmax over singleton axis == 1), so
// the fc1/fc2 weights never touch the output.
// ---------------------------------------------------------------------------
__global__ void combine_init(const float* __restrict__ acc, const float* __restrict__ xw,
                             const float* __restrict__ snorm, const float* __restrict__ bias,
                             float* __restrict__ out,
                             unsigned short* __restrict__ prevbf,
                             unsigned short* __restrict__ xchbf) {
    long long i = (long long)blockIdx.x * blockDim.x + threadIdx.x;
    if (i >= (long long)kN * kD) return;
    int n = (int)(i / kD), j = (int)(i % kD);
    float v = acc[i] + xw[i] * snorm[n] + bias[j];
    v = leaky2(v);
    out[(long long)n * (kL * kD) + j] = v;                       // outs[0]
    prevbf[i] = f2bf(v);
    xchbf[(long long)n * kD + (j % kK) * kC + (j / kK)] = f2bf(v);
}

__global__ void combine_zl(const float* __restrict__ acc, float* __restrict__ xw,
                           const float* __restrict__ snorm, const float* __restrict__ bias) {
    long long i = (long long)blockIdx.x * blockDim.x + threadIdx.x;
    if (i >= (long long)kN * kD) return;
    int n = (int)(i / kD), j = (int)(i % kD);
    xw[i] = leaky2(acc[i] + xw[i] * snorm[n] + bias[j]);         // zl (gate == id)
}

__global__ void combine_res(const float* __restrict__ acc, float* __restrict__ xw,
                            const float* __restrict__ snorm, const float* __restrict__ bias) {
    long long i = (long long)blockIdx.x * blockDim.x + threadIdx.x;
    if (i >= (long long)kN * kD) return;
    int n = (int)(i / kD), j = (int)(i % kD);
    xw[i] = acc[i] + xw[i] * snorm[n] + bias[j];                 // pre-softmax res
}

// Row softmax of rr + add zl, write output block, emit next-layer activations.
__global__ void softmax_add_kernel(const float* __restrict__ rr, const float* __restrict__ zl,
                                   float* __restrict__ out, int colBase,
                                   unsigned short* __restrict__ prevbf,
                                   unsigned short* __restrict__ xchbf) {
    int n = blockIdx.x;
    const float* r = rr + (long long)n * kD;
    const float* z = zl + (long long)n * kD;
    __shared__ float sbuf[256];
    int t = threadIdx.x;

    float mx = -INFINITY;
    for (int j = t; j < kD; j += 256) mx = fmaxf(mx, r[j]);
    sbuf[t] = mx; __syncthreads();
    for (int s = 128; s > 0; s >>= 1) {
        if (t < s) sbuf[t] = fmaxf(sbuf[t], sbuf[t + s]);
        __syncthreads();
    }
    mx = sbuf[0]; __syncthreads();

    float sum = 0.0f;
    for (int j = t; j < kD; j += 256) sum += __expf(r[j] - mx);
    sbuf[t] = sum; __syncthreads();
    for (int s = 128; s > 0; s >>= 1) {
        if (t < s) sbuf[t] += sbuf[t + s];
        __syncthreads();
    }
    float inv = 1.0f / sbuf[0];

    for (int j = t; j < kD; j += 256) {
        float v = z[j] + __expf(r[j] - mx) * inv;
        out[(long long)n * (kL * kD) + colBase + j] = v;
        prevbf[(long long)n * kD + j] = f2bf(v);
        xchbf[(long long)n * kD + (j % kK) * kC + (j / kK)] = f2bf(v);
    }
}

// ---------------------------------------------------------------------------
// Host orchestration
// ---------------------------------------------------------------------------
static inline char* bump(char*& p, long long bytes) {
    char* r = p;
    p += (bytes + 255) & ~255LL;
    return r;
}

extern "C" void kernel_launch(void* const* d_in, const int* in_sizes, int n_in,
                              void* d_out, int out_size, void* d_ws, size_t ws_size,
                              hipStream_t stream) {
    (void)in_sizes; (void)n_in; (void)out_size; (void)ws_size;

    const float* x       = (const float*)d_in[0];
    const int*   eidx    = (const int*)  d_in[1];
    const float* W_init  = (const float*)d_in[2];
    const float* b_init  = (const float*)d_in[3];
    // d_in[4], d_in[5]: fc1_0/fc2_0 — gate is mathematically the identity, unused
    const float* W_conv  = (const float*)d_in[6];
    const float* b_conv  = (const float*)d_in[7];
    // d_in[8], d_in[9]: fc1_l/fc2_l — unused (gate identity)
    const float* W_res   = (const float*)d_in[10];
    const float* b_res   = (const float*)d_in[11];
    float* out = (float*)d_out;

    const int* src = eidx;
    const int* dst = eidx + kE;

    // Workspace carve-up
    char* p = (char*)d_ws;
    float* deg    = (float*)bump(p, sizeof(float) * kN);
    float* dinv   = (float*)bump(p, sizeof(float) * kN);
    float* snorm  = (float*)bump(p, sizeof(float) * kN);
    float* enorm  = (float*)bump(p, sizeof(float) * kE);
    unsigned short* xbf    = (unsigned short*)bump(p, 2LL * kN * kCin);
    unsigned short* wcatT  = (unsigned short*)bump(p, 2LL * kD * kCin);
    unsigned short* wconvT = (unsigned short*)bump(p, 2LL * (kL - 1) * kK * kC * kC);
    unsigned short* wresT  = (unsigned short*)bump(p, 2LL * (kL - 1) * kD * kD);
    float* gemmA  = (float*)bump(p, sizeof(float) * (long long)kN * kD);
    float* gemmB  = (float*)bump(p, sizeof(float) * (long long)kN * kD);
    float* acc1   = (float*)bump(p, sizeof(float) * (long long)kN * kD);
    float* acc2   = (float*)bump(p, sizeof(float) * (long long)kN * kD);
    unsigned short* prevbf = (unsigned short*)bump(p, 2LL * kN * kD);
    unsigned short* xchbf  = (unsigned short*)bump(p, 2LL * kN * kD);

    const long long ND = (long long)kN * kD;
    auto blk = [](long long n) { return (unsigned)((n + 255) / 256); };

    // ---- GCN normalization coefficients ----
    zero_f32<<<blk(kN), 256, 0, stream>>>(deg, kN);
    deg_kernel<<<blk(kE), 256, 0, stream>>>(dst, deg);
    norm_kernel<<<blk(kN), 256, 0, stream>>>(deg, dinv, snorm);
    enorm_kernel<<<blk(kE), 256, 0, stream>>>(src, dst, dinv, enorm);

    // ---- bf16 conversions ----
    cvt_bf16<<<blk((long long)kN * kCin), 256, 0, stream>>>(x, xbf, (long long)kN * kCin);
    build_wcatT <<<blk((long long)kD * kCin), 256, 0, stream>>>(W_init, wcatT);
    build_wconvT<<<blk((long long)(kL - 1) * kK * kC * kC), 256, 0, stream>>>(W_conv, wconvT);
    build_wresT <<<blk((long long)(kL - 1) * kD * kD), 256, 0, stream>>>(W_res, wresT);

    const int Mt = kN / 16;   // 625

    // ---- initial block: xw = x @ Wcat  (10000x512 @ 512x1024) ----
    // 16 strips of 64 cols; 4 waves/block -> grid.y = 4
    wmma_gemm_bf16<<<dim3(Mt, 4, 1), 128, 0, stream>>>(
        xbf, kCin, 0, wcatT, kCin, 0, gemmA, kD, 0, kCin);
    zero_f32<<<blk(ND), 256, 0, stream>>>(acc1, ND);
    scatter_kernel<<<kE, 256, 0, stream>>>(src, dst, enorm, gemmA, acc1);
    combine_init<<<blk(ND), 256, 0, stream>>>(acc1, gemmA, snorm, b_init, out, prevbf, xchbf);

    // ---- L-1 LayerGT blocks ----
    for (int l = 0; l < kL - 1; ++l) {
        // block-diagonal conv: K batches of 10000x128 @ 128x128
        // 2 strips of 64 cols; 2 waves/block -> grid.y = 1, blockDim 64
        wmma_gemm_bf16<<<dim3(Mt, 1, kK), 64, 0, stream>>>(
            xchbf, kD, kC,
            wconvT + (long long)l * kK * kC * kC, kC, (long long)kC * kC,
            gemmA, kD, kC,
            kC);
        // residual GCN transform: 10000x1024 @ 1024x1024
        wmma_gemm_bf16<<<dim3(Mt, 4, 1), 128, 0, stream>>>(
            prevbf, kD, 0,
            wresT + (long long)l * kD * kD, kD, 0,
            gemmB, kD, 0,
            kD);

        zero_f32<<<blk(ND), 256, 0, stream>>>(acc1, ND);
        scatter_kernel<<<kE, 256, 0, stream>>>(src, dst, enorm, gemmA, acc1);
        zero_f32<<<blk(ND), 256, 0, stream>>>(acc2, ND);
        scatter_kernel<<<kE, 256, 0, stream>>>(src, dst, enorm, gemmB, acc2);

        combine_zl <<<blk(ND), 256, 0, stream>>>(acc1, gemmA, snorm, b_conv + (long long)l * kD);
        combine_res<<<blk(ND), 256, 0, stream>>>(acc2, gemmB, snorm, b_res  + (long long)l * kD);

        softmax_add_kernel<<<kN, 256, 0, stream>>>(gemmB, gemmA, out, (l + 1) * kD,
                                                   prevbf, xchbf);
    }
}
